// MinimalCopresheafTNN_23691039605496
// MI455X (gfx1250) — matrix operand order repaired and checked by reading, not compiled, not run on hardware
//
#include <hip/hip_runtime.h>
#include <stdint.h>

#define DCH 128

typedef __attribute__((ext_vector_type(16))) __bf16 v16bf;
typedef __attribute__((ext_vector_type(8)))  float  v8f;

static __device__ __forceinline__ unsigned f2bf_bits(float f) {
  unsigned u = __builtin_bit_cast(unsigned, f);
  return (u + 0x7FFFu + ((u >> 16) & 1u)) >> 16;
}
static __device__ __forceinline__ __bf16 bf_from_bits(unsigned short s) {
  return __builtin_bit_cast(__bf16, s);
}

static __device__ __forceinline__ v8f wmma_bf16(v16bf a, v16bf b, v8f c) {
  // D = A(16x32 bf16) * B(32x16 bf16) + C(16x16 f32)
  return __builtin_amdgcn_wmma_f32_16x16x32_bf16(false, a, false, b, (short)0, c, false, false);
}

// row mean / inv-std over 128 contiguous floats
static __device__ __forceinline__ void rowstats(const float* r, float* mu, float* rsig) {
  float s = 0.f, s2 = 0.f;
#pragma unroll
  for (int i = 0; i < DCH; i += 4) {
    float4 v = *(const float4*)(r + i);
    s  += v.x + v.y + v.z + v.w;
    s2 += v.x*v.x + v.y*v.y + v.z*v.z + v.w*v.w;
  }
  float mean = s * (1.0f / DCH);
  float var  = s2 * (1.0f / DCH) - mean * mean;
  *mu = mean;
  *rsig = rsqrtf(var + 1e-5f);
}

// ---------------- prep kernels ----------------

__global__ void k_hist(const int* __restrict__ pols, unsigned* __restrict__ cnt, int N, int P) {
  int i = blockIdx.x * 256 + threadIdx.x;
  if (i < N) { int p = pols[i] % P; if (p < 0) p += P; atomicAdd(&cnt[p], 1u); }
}

__global__ void k_deg(const int* __restrict__ row, unsigned* __restrict__ deg, int E) {
  int i = blockIdx.x * 256 + threadIdx.x;
  if (i < E) atomicAdd(&deg[row[i]], 1u);
}

__global__ void k_scan(const unsigned* __restrict__ cnt, unsigned* __restrict__ cur, int P) {
  if (threadIdx.x == 0 && blockIdx.x == 0) {
    unsigned o = 0;
    for (int p = 0; p < P; ++p) { cur[p] = o; o += ((cnt[p] + 15u) >> 4) << 4; }
  }
}

__global__ void k_scatter(const int* __restrict__ pols, unsigned* __restrict__ cur,
                          int* __restrict__ order, int N, int P) {
  int i = blockIdx.x * 256 + threadIdx.x;
  if (i < N) {
    int p = pols[i] % P; if (p < 0) p += P;
    unsigned s = atomicAdd(&cur[p], 1u);
    order[s] = i;
  }
}

// pack index layout: i = ((p*4 + kt)*8 + nt)*512 + lane*16 + e
// B-frag (32x16 bf16): lane<16 -> K = kt*32 + (e<8 ? e : e+8),  N = nt*16 + lane
//                      lane>=16 -> K = kt*32+8+(e<8 ? e : e+8), N = nt*16 + lane-16
__global__ void k_spack(const float* __restrict__ send, const float* __restrict__ dsend,
                        __bf16* __restrict__ spack, int P) {
  int i = blockIdx.x * 256 + threadIdx.x;
  if (i >= P * 16384) return;
  int e = i & 15, lane = (i >> 4) & 31, nt = (i >> 9) & 7, kt = (i >> 12) & 3, p = i >> 14;
  int k = kt * 32 + ((lane & 16) ? 8 : 0) + (e < 8 ? e : 8 + e);
  int n = nt * 16 + (lane & 15);
  long long si = (long long)p * 16384 + k * DCH + n;   // einsum 'nd,pde': B[d][e] = S[p][d][e]
  spack[i] = (__bf16)(send[si] + dsend[si]);
}

// M_p[d][e] = sum_j W_r[j][d] * (recv+drecv)[p][j][e]   (W_r^T folded into receive map)
__global__ void k_mpack(const float* __restrict__ wr, const float* __restrict__ recv,
                        const float* __restrict__ drecv, __bf16* __restrict__ mpack, int P) {
  int i = blockIdx.x * 256 + threadIdx.x;
  if (i >= P * 16384) return;
  int e = i & 15, lane = (i >> 4) & 31, nt = (i >> 9) & 7, kt = (i >> 12) & 3, p = i >> 14;
  int k = kt * 32 + ((lane & 16) ? 8 : 0) + (e < 8 ? e : 8 + e);
  int n = nt * 16 + (lane & 15);
  long long base = (long long)p * 16384;
  float s = 0.f;
  for (int j = 0; j < DCH; ++j)
    s += wr[j * DCH + k] * (recv[base + j * DCH + n] + drecv[base + j * DCH + n]);
  mpack[i] = (__bf16)s;
}

// B[k][n] = W[n][k]  (for X @ W^T)
__global__ void k_wpack(const float* __restrict__ w, __bf16* __restrict__ pack) {
  int i = blockIdx.x * 256 + threadIdx.x;
  if (i >= 16384) return;
  int e = i & 15, lane = (i >> 4) & 31, nt = (i >> 9) & 7, kt = (i >> 12) & 3;
  int k = kt * 32 + ((lane & 16) ? 8 : 0) + (e < 8 ? e : 8 + e);
  int n = nt * 16 + (lane & 15);
  pack[i] = (__bf16)w[n * DCH + k];
}

// ---------------- phase 1: x_send = x @ S_p (one wave per 16-node tile) ----------------

__global__ __launch_bounds__(128) void k_send(
    const float* __restrict__ x, const __bf16* __restrict__ spack,
    const int* __restrict__ order, const int* __restrict__ pols,
    unsigned* __restrict__ xsend, int ntiles, int P)
{
  __shared__ float tile[4][16 * DCH];
  __shared__ int   snode[4][16];
  int wave = threadIdx.x >> 5, lane = threadIdx.x & 31;
  int t = blockIdx.x * 4 + wave;
  if (t >= ntiles) return;
  int first = order[(long long)t * 16];
  if (first < 0) return;                        // fully past end
  int p = pols[first] % P; if (p < 0) p += P;   // uniform polarity per tile (padded buckets)
  int m = lane & 15;
  int node = order[(long long)t * 16 + m];
  if (lane < 16) snode[wave][lane] = node;
  long long rb = (node >= 0) ? (long long)node * DCH : 0;

  v8f acc[8] = {};
  const __bf16* bp = spack + (long long)p * (4 * 8 * 512);
#pragma unroll
  for (int kt = 0; kt < 4; ++kt) {
    // A-frag 16x32 bf16: lane<16 holds K = kt*32 + {0..7,16..23}; lane>=16: +8
    const float* ar = x + rb + kt * 32 + ((lane & 16) ? 8 : 0);
    v16bf a;
#pragma unroll
    for (int i = 0; i < 8; ++i) a[i] = (__bf16)ar[i];
#pragma unroll
    for (int i = 0; i < 8; ++i) a[8 + i] = (__bf16)ar[16 + i];
#pragma unroll
    for (int nt = 0; nt < 8; ++nt) {
      v16bf b = *(const v16bf*)(bp + ((kt * 8 + nt) * 512 + lane * 16));
      acc[nt] = wmma_bf16(a, b, acc[nt]);
    }
  }
  // C layout: vgpr i -> row = i + (lane&16 ? 8:0), col = nt*16 + (lane&15)
  float* tl = tile[wave];
  int crow = (lane & 16) ? 8 : 0, ccol = (lane & 15);
#pragma unroll
  for (int nt = 0; nt < 8; ++nt)
#pragma unroll
    for (int i = 0; i < 8; ++i)
      tl[(crow + i) * DCH + nt * 16 + ccol] = acc[nt][i];

  // pack bf16 pairs, scatter rows to xsend
  for (int idx = lane; idx < 16 * 64; idx += 32) {
    int r = idx >> 6, cp = idx & 63;
    int nd = snode[wave][r];
    if (nd >= 0) {
      unsigned lo = f2bf_bits(tl[r * DCH + 2 * cp]);
      unsigned hi = f2bf_bits(tl[r * DCH + 2 * cp + 1]);
      xsend[(long long)nd * 64 + cp] = lo | (hi << 16);
    }
  }
}

// ---------------- phase 2: edge gather + bf16 packed atomic scatter ----------------

__global__ __launch_bounds__(256) void k_edge(
    const unsigned* __restrict__ xsend, const int* __restrict__ row,
    const int* __restrict__ col, unsigned* __restrict__ agg, int E)
{
  long long i = (long long)blockIdx.x * 256 + threadIdx.x;
  if (i >= (long long)E * 16) return;
  int e  = (int)(i >> 4);
  int c4 = (int)(i & 15) * 4;          // 4 dwords = 8 bf16 channels
  int r = row[e], c = col[e];
  uint4 v = *(const uint4*)(xsend + (long long)r * 64 + c4);
  unsigned long long dst = (unsigned long long)(agg + (long long)c * 64 + c4);
  asm volatile("global_atomic_pk_add_bf16 %0, %1, off\n\t"
               "global_atomic_pk_add_bf16 %0, %2, off offset:4\n\t"
               "global_atomic_pk_add_bf16 %0, %3, off offset:8\n\t"
               "global_atomic_pk_add_bf16 %0, %4, off offset:12"
               :: "v"(dst), "v"(v.x), "v"(v.y), "v"(v.z), "v"(v.w) : "memory");
}

// ------- phase 3 (fused): receive map, deg-norm, MLP, LN, ReLU, residual, final LN -------

__global__ __launch_bounds__(128) void k_update(
    const unsigned* __restrict__ agg, const __bf16* __restrict__ mpack,
    const __bf16* __restrict__ w1pack, const __bf16* __restrict__ w2pack,
    const float* __restrict__ x,
    const float* __restrict__ b1, const float* __restrict__ g1, const float* __restrict__ bln1,
    const float* __restrict__ b2, const float* __restrict__ ng, const float* __restrict__ nb,
    const float* __restrict__ rscale,
    const int* __restrict__ order, const int* __restrict__ pols,
    const unsigned* __restrict__ deg,
    float* __restrict__ out, int ntiles, int P)
{
  __shared__ float tile[4][16 * DCH];
  __shared__ int   snode[4][16];
  __shared__ float sdn[4][16];
  __shared__ float smu[4][16], srs[4][16];
  __shared__ float sp[6][DCH]; // 0:b1 1:ln1_g 2:ln1_b 3:b2 4:norm_g 5:norm_b
  for (int i = threadIdx.x; i < DCH; i += 128) {
    sp[0][i] = b1[i]; sp[1][i] = g1[i]; sp[2][i] = bln1[i];
    sp[3][i] = b2[i]; sp[4][i] = ng[i]; sp[5][i] = nb[i];
  }
  __syncthreads();

  float rs = rscale[0];
  int wave = threadIdx.x >> 5, lane = threadIdx.x & 31;
  int t = blockIdx.x * 4 + wave;
  if (t >= ntiles) return;
  int first = order[(long long)t * 16];
  if (first < 0) return;
  int p = pols[first] % P; if (p < 0) p += P;
  int m = lane & 15;
  int node = order[(long long)t * 16 + m];
  if (lane < 16) {
    snode[wave][lane] = node;
    unsigned d = (node >= 0) ? deg[node] : 1u;
    sdn[wave][lane] = 1.0f / (float)(d > 0 ? d : 1u);
  }
  long long rb64 = (node >= 0) ? (long long)node * 64 : 0;
  float* tl = tile[wave];
  int crow = (lane & 16) ? 8 : 0, ccol = (lane & 15);

  // ---- out = deg_norm * (agg0 @ M_p) ----
  {
    v8f acc[8] = {};
    const __bf16* bp = mpack + (long long)p * (4 * 8 * 512);
#pragma unroll
    for (int kt = 0; kt < 4; ++kt) {
      const unsigned* a0 = agg + rb64 + kt * 16 + ((lane & 16) ? 4 : 0);
      v16bf a;
#pragma unroll
      for (int i = 0; i < 4; ++i) {
        unsigned u = a0[i];
        a[2 * i]     = bf_from_bits((unsigned short)(u & 0xffffu));
        a[2 * i + 1] = bf_from_bits((unsigned short)(u >> 16));
      }
#pragma unroll
      for (int i = 0; i < 4; ++i) {
        unsigned u = a0[8 + i];
        a[8 + 2 * i] = bf_from_bits((unsigned short)(u & 0xffffu));
        a[9 + 2 * i] = bf_from_bits((unsigned short)(u >> 16));
      }
#pragma unroll
      for (int nt = 0; nt < 8; ++nt) {
        v16bf b = *(const v16bf*)(bp + ((kt * 8 + nt) * 512 + lane * 16));
        acc[nt] = wmma_bf16(a, b, acc[nt]);
      }
    }
#pragma unroll
    for (int nt = 0; nt < 8; ++nt)
#pragma unroll
      for (int i = 0; i < 8; ++i) {
        int r = crow + i;
        tl[r * DCH + nt * 16 + ccol] = acc[nt][i] * sdn[wave][r];
      }
  }

  // ---- h1pre = out @ W1^T + b1 ----
  {
    v8f acc[8] = {};
#pragma unroll
    for (int kt = 0; kt < 4; ++kt) {
      const float* ar = tl + m * DCH + kt * 32 + ((lane & 16) ? 8 : 0);
      v16bf a;
#pragma unroll
      for (int i = 0; i < 8; ++i) a[i] = (__bf16)ar[i];
#pragma unroll
      for (int i = 0; i < 8; ++i) a[8 + i] = (__bf16)ar[16 + i];
#pragma unroll
      for (int nt = 0; nt < 8; ++nt) {
        v16bf b = *(const v16bf*)(w1pack + ((kt * 8 + nt) * 512 + lane * 16));
        acc[nt] = wmma_bf16(a, b, acc[nt]);
      }
    }
#pragma unroll
    for (int nt = 0; nt < 8; ++nt)
#pragma unroll
      for (int i = 0; i < 8; ++i) {
        int r = crow + i, c = nt * 16 + ccol;
        tl[r * DCH + c] = acc[nt][i] + sp[0][c];
      }
  }

  // ---- h2 = relu(LN1(h1pre)) @ W2^T ----
  float mu, rsig;
  rowstats(tl + m * DCH, &mu, &rsig);
  {
    v8f acc[8] = {};
#pragma unroll
    for (int kt = 0; kt < 4; ++kt) {
      int kb = kt * 32 + ((lane & 16) ? 8 : 0);
      const float* ar = tl + m * DCH + kb;
      v16bf a;
#pragma unroll
      for (int i = 0; i < 8; ++i) {
        int c = kb + i;
        float v = (ar[i] - mu) * rsig * sp[1][c] + sp[2][c];
        a[i] = (__bf16)(v > 0.f ? v : 0.f);
      }
#pragma unroll
      for (int i = 0; i < 8; ++i) {
        int c = kb + 16 + i;
        float v = (ar[16 + i] - mu) * rsig * sp[1][c] + sp[2][c];
        a[8 + i] = (__bf16)(v > 0.f ? v : 0.f);
      }
#pragma unroll
      for (int nt = 0; nt < 8; ++nt) {
        v16bf b = *(const v16bf*)(w2pack + ((kt * 8 + nt) * 512 + lane * 16));
        acc[nt] = wmma_bf16(a, b, acc[nt]);
      }
    }
    // y = res_scale*(h2 + b2) + x
#pragma unroll
    for (int nt = 0; nt < 8; ++nt)
#pragma unroll
      for (int i = 0; i < 8; ++i) {
        int r = crow + i, c = nt * 16 + ccol;
        int nd = snode[wave][r];
        float xv = (nd >= 0) ? x[(long long)nd * DCH + c] : 0.f;
        tl[r * DCH + c] = rs * (acc[nt][i] + sp[3][c]) + xv;
      }
  }

  // ---- final LN -> d_out ----
  float mu2, rs2;
  rowstats(tl + m * DCH, &mu2, &rs2);
  if (lane < 16) { smu[wave][lane] = mu2; srs[wave][lane] = rs2; }
  for (int idx = lane; idx < 16 * DCH; idx += 32) {
    int r = idx >> 7, c = idx & 127;
    int nd = snode[wave][r];
    if (nd >= 0)
      out[(long long)nd * DCH + c] = (tl[idx] - smu[wave][r]) * srs[wave][r] * sp[4][c] + sp[5][c];
  }
}

// ---------------- launch ----------------

extern "C" void kernel_launch(void* const* d_in, const int* in_sizes, int n_in,
                              void* d_out, int out_size, void* d_ws, size_t ws_size,
                              hipStream_t stream)
{
  const float* x     = (const float*)d_in[0];
  const float* send  = (const float*)d_in[1];
  const float* recv  = (const float*)d_in[2];
  const float* dsend = (const float*)d_in[3];
  const float* drecv = (const float*)d_in[4];
  const float* wr    = (const float*)d_in[5];
  const float* w1    = (const float*)d_in[6];
  const float* b1    = (const float*)d_in[7];
  const float* g1    = (const float*)d_in[8];
  const float* bln1  = (const float*)d_in[9];
  const float* w2    = (const float*)d_in[10];
  const float* b2    = (const float*)d_in[11];
  const float* ng    = (const float*)d_in[12];
  const float* nb    = (const float*)d_in[13];
  const float* rsc   = (const float*)d_in[14];
  const int*   row   = (const int*)d_in[15];
  const int*   col   = (const int*)d_in[16];
  const int*   pols  = (const int*)d_in[17];

  int N = in_sizes[0] / DCH;
  int P = in_sizes[1] / (DCH * DCH);
  int E = in_sizes[15];
  int ntiles = (N + 15) / 16 + P;   // padded-bucket upper bound

  uint8_t* ws = (uint8_t*)d_ws;
  size_t o = 0;
  auto alloc = [&](size_t bytes) { size_t r = o; o = (o + bytes + 255) & ~(size_t)255; return r; };
  unsigned* xsend  = (unsigned*)(ws + alloc((size_t)N * 64 * 4));
  unsigned* agg    = (unsigned*)(ws + alloc((size_t)N * 64 * 4));
  unsigned* deg    = (unsigned*)(ws + alloc((size_t)N * 4));
  int*      order  = (int*)(ws + alloc((size_t)ntiles * 16 * 4));
  unsigned* cnt    = (unsigned*)(ws + alloc(64));
  unsigned* cur    = (unsigned*)(ws + alloc(64));
  __bf16*   spack  = (__bf16*)(ws + alloc((size_t)P * 16384 * 2));
  __bf16*   mpack  = (__bf16*)(ws + alloc((size_t)P * 16384 * 2));
  __bf16*   w1pack = (__bf16*)(ws + alloc(16384 * 2));
  __bf16*   w2pack = (__bf16*)(ws + alloc(16384 * 2));

  (void)hipMemsetAsync(agg, 0, (size_t)N * 64 * 4, stream);
  (void)hipMemsetAsync(deg, 0, (size_t)N * 4, stream);
  (void)hipMemsetAsync(cnt, 0, 64, stream);
  (void)hipMemsetAsync(order, 0xFF, (size_t)ntiles * 16 * 4, stream);  // -1 fill

  const int tb = 256;
  k_hist<<<(N + tb - 1) / tb, tb, 0, stream>>>(pols, cnt, N, P);
  k_deg<<<(E + tb - 1) / tb, tb, 0, stream>>>(row, deg, E);
  k_scan<<<1, 64, 0, stream>>>(cnt, cur, P);
  k_scatter<<<(N + tb - 1) / tb, tb, 0, stream>>>(pols, cur, order, N, P);
  int ptot = P * 16384;
  k_spack<<<(ptot + tb - 1) / tb, tb, 0, stream>>>(send, dsend, spack, P);
  k_mpack<<<(ptot + tb - 1) / tb, tb, 0, stream>>>(wr, recv, drecv, mpack, P);
  k_wpack<<<(16384 + tb - 1) / tb, tb, 0, stream>>>(w1, w1pack);
  k_wpack<<<(16384 + tb - 1) / tb, tb, 0, stream>>>(w2, w2pack);

  k_send<<<(ntiles + 3) / 4, 128, 0, stream>>>(x, spack, order, pols, xsend, ntiles, P);

  long long ework = (long long)E * 16;
  k_edge<<<(unsigned)((ework + tb - 1) / tb), tb, 0, stream>>>(xsend, row, col, agg, E);

  k_update<<<(ntiles + 3) / 4, 128, 0, stream>>>(agg, mpack, w1pack, w2pack, x,
      b1, g1, bln1, b2, ng, nb, rsc, order, pols, deg, (float*)d_out, ntiles, P);
}
